// SelfAttention_v1_70282844832522
// MI455X (gfx1250) — matrix operand order repaired and checked by reading, not compiled, not run on hardware
//
#include <hip/hip_runtime.h>
#include <hip/hip_bf16.h>

// Self-attention (N=8192, D_IN=512, D_OUT=64) for gfx1250:
//   1) weight transpose + f32->bf16
//   2) QKV projection via v_wmma_f32_16x16x32_bf16, software-pipelined B loads
//      (Q pre-scaled by 1/sqrt(64) * log2(e) -> exp2-domain softmax)
//   3) flash-attention, 32 queries per wave; V fragments loaded early to
//      overlap softmax; denominator via extra WMMA against a ones-column.

#define N_SEQ 8192
#define D_IN  512
#define D_OUT 64

typedef __attribute__((ext_vector_type(16))) __bf16 v16bf;
typedef __attribute__((ext_vector_type(8)))  float  v8f;

union BF16x16 {
  v16bf v;
  unsigned int w[8];
  unsigned short u[16];
  uint4 q[2];
};

// f32 -> bf16, round-half-up (1-2 VALU ops instead of the 4-op RNE chain).
__device__ __forceinline__ unsigned short f2bf(float f) {
  return (unsigned short)((__float_as_uint(f) + 0x8000u) >> 16);
}

// Two f32 -> packed bf16x2: one v_add per input + one v_perm_b32.
__device__ __forceinline__ unsigned int f2bf_pk(float lo, float hi) {
  unsigned int a = __float_as_uint(lo) + 0x8000u;
  unsigned int b = __float_as_uint(hi) + 0x8000u;
#if __has_builtin(__builtin_amdgcn_perm)
  return __builtin_amdgcn_perm(b, a, 0x07060302u);   // {b[31:16], a[31:16]}
#else
  return (a >> 16) | (b & 0xFFFF0000u);
#endif
}

__device__ __forceinline__ v8f wmma_bf16(v16bf a, v16bf b, v8f c) {
  // (neg_a, A, neg_b, B, c_mod, C, reuse_a, reuse_b)
  return __builtin_amdgcn_wmma_f32_16x16x32_bf16(false, a, false, b, (short)0, c,
                                                 false, false);
}

// B fragment (32x16 = K x N): B[k][n] = M[(n0+n)*ld + k0 + k].
// Lane n = lane&15 holds k = base..base+7 and base+16..base+23 with base = (lane>>4)*8.
__device__ __forceinline__ v16bf load_bfrag(const unsigned short* M, int ld,
                                            int n0, int k0, int lane) {
  const unsigned short* p = M + (size_t)(n0 + (lane & 15)) * ld + k0 + ((lane >> 4) << 3);
  BF16x16 r;
  r.q[0] = *(const uint4*)p;
  r.q[1] = *(const uint4*)(p + 16);
  return r.v;
}

// A fragment (16x32 = M x K): lane holds row = lane&15,
// k = base..base+7 and base+16..base+23 with base = (lane>>4)*8.
__device__ __forceinline__ v16bf load_afrag(const unsigned short* M, int ld,
                                            int m0, int k0, int lane) {
  const unsigned short* p = M + (size_t)(m0 + (lane & 15)) * ld + k0 + ((lane >> 4) << 3);
  BF16x16 r;
  r.q[0] = *(const uint4*)p;
  r.q[1] = *(const uint4*)(p + 16);
  return r.v;
}

// --------------------------------------------------------------------------
// Kernel 0: W[512][64] f32  ->  WT[64][512] bf16 (x3 matrices)
// --------------------------------------------------------------------------
__global__ __launch_bounds__(128)
void wt_cvt_kernel(const float* __restrict__ Wq, const float* __restrict__ Wk,
                   const float* __restrict__ Wv,
                   unsigned short* __restrict__ WqT, unsigned short* __restrict__ WkT,
                   unsigned short* __restrict__ WvT) {
  const float* W = (blockIdx.y == 0) ? Wq : (blockIdx.y == 1) ? Wk : Wv;
  unsigned short* WT = (blockIdx.y == 0) ? WqT : (blockIdx.y == 1) ? WkT : WvT;
  const int o = blockIdx.x;                      // output dim 0..63
  for (int i = threadIdx.x; i < D_IN; i += 128)
    WT[o * D_IN + i] = f2bf(W[i * D_OUT + o]);
}

// --------------------------------------------------------------------------
// Kernel 1: QKV projection. One wave handles a 16-row block of x.
//   Q (scaled by log2(e)/8) and K stored row-major bf16 [N][64];
//   V stored transposed bf16 [64][N] so attention B-fragments are contiguous.
//   B fragments double-buffered (depth 2) so WMMAs overlap L2 loads.
// --------------------------------------------------------------------------
__global__ __launch_bounds__(32)
void qkv_kernel(const float* __restrict__ x,
                const unsigned short* __restrict__ WqT,
                const unsigned short* __restrict__ WkT,
                const unsigned short* __restrict__ WvT,
                unsigned short* __restrict__ Qb,
                unsigned short* __restrict__ Kb,
                unsigned short* __restrict__ VTb) {
  const int lane = threadIdx.x;
  const int row0 = blockIdx.x * 16;
  const int half = lane >> 4;    // 0 or 1
  const int nl   = lane & 15;

  // Load whole 16x512 x block once, as 16 bf16 A fragments (128 VGPRs).
  v16bf a[16];
  {
    const float* xr = x + (size_t)(row0 + nl) * D_IN + (half << 3);
    #pragma unroll
    for (int kb = 0; kb < 16; ++kb) {
      const float* p = xr + kb * 32;
      float4 f0 = *(const float4*)(p + 0);
      float4 f1 = *(const float4*)(p + 4);
      float4 f2 = *(const float4*)(p + 16);
      float4 f3 = *(const float4*)(p + 20);
      BF16x16 r;
      r.w[0] = f2bf_pk(f0.x, f0.y);  r.w[1] = f2bf_pk(f0.z, f0.w);
      r.w[2] = f2bf_pk(f1.x, f1.y);  r.w[3] = f2bf_pk(f1.z, f1.w);
      r.w[4] = f2bf_pk(f2.x, f2.y);  r.w[5] = f2bf_pk(f2.z, f2.w);
      r.w[6] = f2bf_pk(f3.x, f3.y);  r.w[7] = f2bf_pk(f3.z, f3.w);
      a[kb] = r.v;
    }
  }

  #pragma unroll
  for (int w = 0; w < 3; ++w) {
    const unsigned short* WT = (w == 0) ? WqT : (w == 1) ? WkT : WvT;
    #pragma unroll
    for (int t = 0; t < 4; ++t) {           // 4 output-dim tiles of 16
      v8f acc = {};
      // depth-2 software pipeline over the 16 K-chunks
      v16bf b0 = load_bfrag(WT, D_IN, t * 16, 0,  lane);
      v16bf b1 = load_bfrag(WT, D_IN, t * 16, 32, lane);
      #pragma unroll
      for (int kb = 0; kb < 16; ++kb) {
        v16bf bn = (kb < 14) ? load_bfrag(WT, D_IN, t * 16, (kb + 2) * 32, lane) : b0;
        acc = wmma_bf16(a[kb], b0, acc);
        b0 = b1;
        b1 = bn;
      }
      if (w == 2) {
        // V transposed: lane writes 8 contiguous keys of one dim row (16B).
        uint4 pk;
        pk.x = f2bf_pk(acc[0], acc[1]);
        pk.y = f2bf_pk(acc[2], acc[3]);
        pk.z = f2bf_pk(acc[4], acc[5]);
        pk.w = f2bf_pk(acc[6], acc[7]);
        *(uint4*)(VTb + (size_t)(t * 16 + nl) * N_SEQ + row0 + (half << 3)) = pk;
      } else {
        // fold 1/sqrt(64) * log2(e) into Q -> softmax runs in exp2 domain
        const float s = (w == 0) ? (0.125f * 1.44269504088896341f) : 1.0f;
        unsigned short* dst = (w == 0) ? Qb : Kb;
        #pragma unroll
        for (int r = 0; r < 8; ++r)
          dst[(size_t)(row0 + r + (half << 3)) * D_OUT + t * 16 + nl] = f2bf(acc[r] * s);
      }
    }
  }
}

// --------------------------------------------------------------------------
// Kernel 2: flash attention, 32 queries (2 row blocks) per wave.
// Per 32-key iteration: 4 K + 4 V B-fragments (V loaded early so softmax
// VALU hides their latency) + 1 constant ones fragment feed 18 WMMAs.
// Softmax denominator = P @ ones accumulated in o[b][4] with the same alpha
// rescaling as O. P re-laid out C->A through 2 KB of LDS.
// --------------------------------------------------------------------------
__global__ __launch_bounds__(32)
void attn_kernel(const unsigned short* __restrict__ Qb,
                 const unsigned short* __restrict__ Kb,
                 const unsigned short* __restrict__ VTb,
                 float* __restrict__ out) {
  __shared__ __align__(16) unsigned short sP[32 * 32];   // 2 KB P tile
  const int lane = threadIdx.x;
  const int half = lane >> 4;
  const int nl   = lane & 15;
  const int q0   = blockIdx.x * 32;

  v16bf qa[2][2];
  #pragma unroll
  for (int b = 0; b < 2; ++b) {
    qa[b][0] = load_afrag(Qb, D_OUT, q0 + b * 16, 0,  lane);
    qa[b][1] = load_afrag(Qb, D_OUT, q0 + b * 16, 32, lane);
  }

  // Constant ones-column B fragment: B[k][0] = 1.0bf16, other columns 0.
  // (lanes 0 and 16 own column n=0 in the 32x16 B layout)
  v16bf vb1;
  {
    const unsigned int ones = (nl == 0) ? 0x3F803F80u : 0u;
    BF16x16 r;
    #pragma unroll
    for (int i = 0; i < 8; ++i) r.w[i] = ones;
    vb1 = r.v;
  }

  v8f o[2][5] = {};            // o[b][4] accumulates the softmax denominator
  float mrow[2][8];
  #pragma unroll
  for (int b = 0; b < 2; ++b)
    #pragma unroll
    for (int r = 0; r < 8; ++r) mrow[b][r] = -1e38f;

  for (int kt = 0; kt < N_SEQ; kt += 32) {
    if (kt + 32 < N_SEQ) {
      __builtin_prefetch(Kb + (size_t)(kt + 32) * D_OUT, 0, 0);    // next K tile
      __builtin_prefetch(VTb + (size_t)(kt + 32), 0, 0);           // next V cols
    }
    // ---- shared K fragments, then V fragments issued early so their L2
    //      latency is hidden behind the S WMMAs + softmax VALU work ----
    const v16bf kb00 = load_bfrag(Kb, D_OUT, kt,      0,  lane);
    const v16bf kb01 = load_bfrag(Kb, D_OUT, kt,      32, lane);
    const v16bf kb10 = load_bfrag(Kb, D_OUT, kt + 16, 0,  lane);
    const v16bf kb11 = load_bfrag(Kb, D_OUT, kt + 16, 32, lane);
    v16bf vbf[4];
    #pragma unroll
    for (int t = 0; t < 4; ++t)
      vbf[t] = load_bfrag(VTb, N_SEQ, t * 16, kt, lane);

    #pragma unroll
    for (int b = 0; b < 2; ++b) {
      // S = Q_b * K^T  (exp2 domain; two 16-col C fragments)
      v8f s0 = {}, s1 = {};
      s0 = wmma_bf16(qa[b][0], kb00, s0);
      s0 = wmma_bf16(qa[b][1], kb01, s0);
      s1 = wmma_bf16(qa[b][0], kb10, s1);
      s1 = wmma_bf16(qa[b][1], kb11, s1);

      // online softmax: only the row-max butterfly remains
      #pragma unroll
      for (int r = 0; r < 8; ++r) {
        float sa = s0[r], sb = s1[r];
        float lm = fmaxf(sa, sb);
        lm = fmaxf(lm, __shfl_xor(lm, 1));
        lm = fmaxf(lm, __shfl_xor(lm, 2));
        lm = fmaxf(lm, __shfl_xor(lm, 4));
        lm = fmaxf(lm, __shfl_xor(lm, 8));
        float mn    = fmaxf(mrow[b][r], lm);
        float alpha = exp2f(mrow[b][r] - mn);
        mrow[b][r] = mn;
        float p0 = exp2f(sa - mn);
        float p1 = exp2f(sb - mn);
        o[b][0][r] *= alpha; o[b][1][r] *= alpha;
        o[b][2][r] *= alpha; o[b][3][r] *= alpha;
        o[b][4][r] *= alpha;
        const int prow = b * 16 + r + (half << 3);   // C layout -> row index
        sP[prow * 32 + nl]      = f2bf(p0);
        sP[prow * 32 + nl + 16] = f2bf(p1);
      }
    }
    __syncthreads();

    // re-read P in A-fragment layout (two ds_load_b128 per lane per block)
    v16bf pa[2];
    #pragma unroll
    for (int b = 0; b < 2; ++b) {
      const unsigned short* p = sP + (b * 16 + nl) * 32 + (half << 3);
      BF16x16 r;
      r.q[0] = *(const uint4*)p;
      r.q[1] = *(const uint4*)(p + 16);
      pa[b] = r.v;
    }

    // ---- O += P * V, denominator += P * 1 (V fragments reused by both) ----
    #pragma unroll
    for (int t = 0; t < 4; ++t) {
      o[0][t] = wmma_bf16(pa[0], vbf[t], o[0][t]);
      o[1][t] = wmma_bf16(pa[1], vbf[t], o[1][t]);
    }
    o[0][4] = wmma_bf16(pa[0], vb1, o[0][4]);
    o[1][4] = wmma_bf16(pa[1], vb1, o[1][4]);
    __syncthreads();
  }

  // ---- finalize: broadcast denominator from column 0, normalize, store ----
  #pragma unroll
  for (int b = 0; b < 2; ++b) {
    #pragma unroll
    for (int r = 0; r < 8; ++r) {
      const float lsum = __shfl(o[b][4][r], half << 4, 32);
      const float inv  = 1.0f / lsum;
      const int grow = q0 + b * 16 + r + (half << 3);
      float* po = out + (size_t)grow * D_OUT + nl;
      po[0]  = o[b][0][r] * inv;
      po[16] = o[b][1][r] * inv;
      po[32] = o[b][2][r] * inv;
      po[48] = o[b][3][r] * inv;
    }
  }
}

// --------------------------------------------------------------------------
extern "C" void kernel_launch(void* const* d_in, const int* in_sizes, int n_in,
                              void* d_out, int out_size, void* d_ws, size_t ws_size,
                              hipStream_t stream) {
  const float* x  = (const float*)d_in[0];
  const float* Wq = (const float*)d_in[1];
  const float* Wk = (const float*)d_in[2];
  const float* Wv = (const float*)d_in[3];
  float* out = (float*)d_out;

  // Workspace layout (bf16 = unsigned short), ~3.4 MB total:
  unsigned short* WqT = (unsigned short*)d_ws;                 // [64][512]
  unsigned short* WkT = WqT + (size_t)D_OUT * D_IN;
  unsigned short* WvT = WkT + (size_t)D_OUT * D_IN;
  unsigned short* Qb  = WvT + (size_t)D_OUT * D_IN;            // [8192][64] (exp2-domain scale)
  unsigned short* Kb  = Qb  + (size_t)N_SEQ * D_OUT;           // [8192][64]
  unsigned short* VTb = Kb  + (size_t)N_SEQ * D_OUT;           // [64][8192]

  wt_cvt_kernel<<<dim3(D_OUT, 3), 128, 0, stream>>>(Wq, Wk, Wv, WqT, WkT, WvT);
  qkv_kernel<<<N_SEQ / 16, 32, 0, stream>>>(x, WqT, WkT, WvT, Qb, Kb, VTb);
  attn_kernel<<<N_SEQ / 32, 32, 0, stream>>>(Qb, Kb, VTb, out);
}
